// kernel_residual_fc_embeds_77206332113313
// MI455X (gfx1250) — matrix operand order repaired and checked by reading, not compiled, not run on hardware
//
#include <hip/hip_runtime.h>

// ---------------------------------------------------------------------------
// MI455X (gfx1250) fused MLP chain:
//   kernel regression -> bf16 WMMA GEMM chain (proj, residual, 6x512x512) -> head
// GEMM: 32x512 block tile, A staged once in LDS, W K-panels double-buffered
// via async global->LDS (ASYNCcnt), compute on v_wmma_f32_16x16x32_bf16.
// ---------------------------------------------------------------------------

typedef __bf16 bf16_t;
typedef __attribute__((ext_vector_type(16))) __bf16 v16bf;
typedef __attribute__((ext_vector_type(8)))  float  v8f;
typedef int v4i_ __attribute__((vector_size(16)));

#define B_    8
#define S_    2000
#define INCH  120
#define GRIDC 64
#define IC    512
#define DIR   4
#define NPTS  10000
#define CATP  256      // 248 padded to 256 (K multiple of 32)
#define NLAY  6
#define MROWS (B_*S_*DIR)   // 64000
#define MBS   (B_*S_)       // 16000

#if __has_builtin(__builtin_amdgcn_global_load_async_to_lds_b128) && \
    __has_builtin(__builtin_amdgcn_s_wait_asynccnt)
#define USE_ASYNC 1
#define WAIT_ASYNC(n) __builtin_amdgcn_s_wait_asynccnt(n)
#else
#define USE_ASYNC 0
#define WAIT_ASYNC(n)
#endif

// Global -> LDS 16B transfer: async (ASYNCcnt path) when available.
__device__ __forceinline__ void g2l_b128(const bf16_t* g, bf16_t* l) {
#if USE_ASYNC
  __builtin_amdgcn_global_load_async_to_lds_b128(
      (__attribute__((address_space(1))) v4i_*)g,
      (__attribute__((address_space(3))) v4i_*)l, 0, 0);
#else
  *(uint4*)l = *(const uint4*)g;
#endif
}

// ---------------------------------------------------------------------------
// 1) Kernel regression: one block per (batch, which-half) query.
//    LDS float atomics (ds_add_f32) for the cross-thread reduction.
// ---------------------------------------------------------------------------
__global__ __launch_bounds__(256) void kr_kernel(
    const float* __restrict__ sound_loc, const float* __restrict__ gcoor,
    const float* __restrict__ xyoff, const float* __restrict__ bws,
    const float* __restrict__ grid0, float* __restrict__ tg)
{
  const int q = blockIdx.x;          // 0..15
  const int b = q >> 1, h = q & 1;
  const float qx = sound_loc[b * 4 + h * 2 + 0];
  const float qy = sound_loc[b * 4 + h * 2 + 1];
  __shared__ float s_acc[65];
  const int tid = threadIdx.x;
  if (tid < 65) s_acc[tid] = 0.0f;
  __syncthreads();

  float acc[GRIDC];
  #pragma unroll
  for (int c = 0; c < GRIDC; ++c) acc[c] = 0.0f;
  float wsum = 0.0f;

  for (int p = tid; p < NPTS; p += 256) {
    const float bx = gcoor[p * 2 + 0] + tanhf(xyoff[p * 2 + 0]) * 0.1f;
    const float by = gcoor[p * 2 + 1] + tanhf(xyoff[p * 2 + 1]) * 0.1f;
    float bw = bws[p];
    bw = fminf(fmaxf(bw, 0.1f), 0.5f);
    const float dx = qx - bx, dy = qy - by;
    const float w = expf(-(dx * dx + dy * dy) / (bw * bw));
    wsum += w;
    const float* g = grid0 + (size_t)p * GRIDC;
    #pragma unroll
    for (int c = 0; c < GRIDC; ++c) acc[c] += w * g[c];
  }
  atomicAdd(&s_acc[64], wsum);
  #pragma unroll
  for (int c = 0; c < GRIDC; ++c) atomicAdd(&s_acc[c], acc[c]);
  __syncthreads();
  if (tid < GRIDC)
    tg[b * (2 * GRIDC) + h * GRIDC + tid] = s_acc[tid] / s_acc[64];
}

// ---------------------------------------------------------------------------
// 2) Build padded bf16 A0 [16000, 256]: [grid(128) | input(120) | pad(8)]
// ---------------------------------------------------------------------------
__global__ __launch_bounds__(256) void build_a0(
    const float* __restrict__ tg, const float* __restrict__ input_stuff,
    bf16_t* __restrict__ A0)
{
  const int idx = blockIdx.x * 256 + threadIdx.x;
  if (idx >= MBS * CATP) return;
  const int m = idx >> 8, c = idx & 255;
  const int b = m / S_;
  float v = 0.0f;
  if (c < 128)      v = tg[b * 128 + c];
  else if (c < 248) v = input_stuff[(size_t)m * INCH + (c - 128)];
  A0[idx] = (bf16_t)v;
}

// ---------------------------------------------------------------------------
// 3) Weight transpose fp32 [K0,512] -> bf16 [512,Kpad], zero K-pad.
// ---------------------------------------------------------------------------
__global__ __launch_bounds__(256) void transpose_w(
    const float* __restrict__ src, int K0, int Kpad, bf16_t* __restrict__ dst)
{
  const int idx = blockIdx.x * 256 + threadIdx.x;
  if (idx >= IC * Kpad) return;
  const int n = idx / Kpad, kk = idx - n * Kpad;
  dst[idx] = (kk < K0) ? (bf16_t)src[(size_t)kk * IC + n] : (bf16_t)0.0f;
}

// ---------------------------------------------------------------------------
// 4) Fold biases + channel embeddings: bias0[d][512], biasK[k][d][512]
// ---------------------------------------------------------------------------
__global__ __launch_bounds__(256) void build_bias(
    const float* __restrict__ proj_b, const float* __restrict__ layers_b,
    const float* __restrict__ channels, float* __restrict__ bias0,
    float* __restrict__ biasK)
{
  const int idx = blockIdx.x * 256 + threadIdx.x;
  if (idx < DIR * IC) {
    const int d = idx >> 9, n = idx & 511;
    bias0[idx] = proj_b[n] + channels[(0 * DIR + d) * IC + n];
  } else if (idx < DIR * IC + NLAY * DIR * IC) {
    const int j = idx - DIR * IC;
    const int k = j / (DIR * IC);
    const int rem = j - k * (DIR * IC);
    const int d = rem >> 9, n = rem & 511;
    biasK[j] = layers_b[k * IC + n] + channels[((k + 1) * DIR + d) * IC + n];
  }
}

// ---------------------------------------------------------------------------
// 5) Generic bf16 WMMA GEMM, N = 512 fixed.
//    Block tile: 32 rows x 512 cols. 8 waves = 2 row-groups x 4 col-groups;
//    each wave owns 16 rows x 128 cols = 8 accumulator tiles (64 VGPRs).
//    A tile (32x520 padded bf16) staged once in LDS -> A read from HBM once.
//    W K-panels (512 x 32) double-buffered in LDS via async global->LDS,
//    prefetched one step ahead and drained with s_wait_asynccnt.
// ---------------------------------------------------------------------------
__global__ __launch_bounds__(256) void gemm512(
    const bf16_t* __restrict__ A, int aDiv4, int K,
    const bf16_t* __restrict__ Wt,
    const float* __restrict__ bias, int biasPerDir,
    const bf16_t* __restrict__ Radd, int leakyIn,
    bf16_t* __restrict__ Y)
{
  __shared__ bf16_t As[32][520];       // 33.3 KB, 260-dword row stride
  __shared__ bf16_t Bs[2][512][40];    // 2 x 40 KB, 20-dword row stride

  const int tid  = threadIdx.x;
  const int wid  = tid >> 5;
  const int lane = tid & 31;
  const int ln   = lane & 15;
  const int hi   = lane >> 4;          // K-half select per ISA A/B layouts
  const int rg   = wid >> 2;           // row group (0..1)
  const int cg   = wid & 3;            // col group (0..3)
  const int rb   = blockIdx.x * 32 + rg * 16;

  // Issue one 512x32 W K-panel into Bs[buf] (8 async b128 per lane).
  auto issuePanel = [&](int k0, int buf) {
    const int n0   = tid >> 2;         // 0..63
    const int kseg = (tid & 3) * 8;    // element offset, 16B aligned
    #pragma unroll
    for (int p = 0; p < 8; ++p) {
      const int n = n0 + p * 64;
      g2l_b128(Wt + (size_t)n * K + k0 + kseg, &Bs[buf][n][kseg]);
    }
  };

  // Prefetch panel 0 while we stage the A tile.
  issuePanel(0, 0);

  // Stage A tile: 32 rows x K, plain coalesced copy (read once from HBM).
  {
    const int cpr = K >> 3;            // 16B chunks per row (32 or 64)
    for (int idx = tid; idx < 32 * cpr; idx += 256) {
      const int r = idx / cpr, c = idx - r * cpr;
      const int gm = blockIdx.x * 32 + r;
      const size_t aRow = (size_t)(aDiv4 ? (gm >> 2) : gm);
      const uint4 v = *(const uint4*)(A + aRow * (size_t)K + c * 8);
      *(uint4*)&As[r][c * 8] = v;
    }
  }
  __syncthreads();

  const v8f vzero = {0.f,0.f,0.f,0.f,0.f,0.f,0.f,0.f};
  v8f acc[8];
  #pragma unroll
  for (int t = 0; t < 8; ++t) acc[t] = vzero;

  const int nsteps = K >> 5;
  for (int s = 0; s < nsteps; ++s) {
    const int k0 = s << 5;
    if (s + 1 < nsteps) {
      issuePanel(k0 + 32, (s + 1) & 1); // prefetch next panel
      WAIT_ASYNC(8);                    // my panel-s transfers complete
    } else {
      WAIT_ASYNC(0);
    }
    __syncthreads();                    // panel s visible to all waves

    // A fragment per ISA 16-bit 16x32 layout (fused leaky-ReLU).
    union AF { uint4 u[2]; v16bf v; __bf16 e[16]; } af;
    {
      const bf16_t* ap = &As[rg * 16 + ln][k0 + hi * 8];
      af.u[0] = *(const uint4*)(ap);
      af.u[1] = *(const uint4*)(ap + 16);
      if (leakyIn) {
        #pragma unroll
        for (int i = 0; i < 16; ++i) {
          const float f = (float)af.e[i];
          af.e[i] = (__bf16)(f > 0.0f ? f : 0.1f * f);
        }
      }
    }

    const int bbuf = s & 1;
    #pragma unroll
    for (int t = 0; t < 8; ++t) {
      union BF { uint4 u[2]; v16bf v; } bfg;
      const bf16_t* bp = &Bs[bbuf][cg * 128 + t * 16 + ln][hi * 8];
      bfg.u[0] = *(const uint4*)(bp);
      bfg.u[1] = *(const uint4*)(bp + 16);
      acc[t] = __builtin_amdgcn_wmma_f32_16x16x32_bf16(
          false, af.v, false, bfg.v, (short)0, acc[t], false, false);
    }
    __syncthreads();                    // reads done before panel overwrite
  }

  // Store per ISA C layout: vgpr r -> M = hi*8 + r, N = ln (+16*tile).
  // rb is a multiple of 16, so dir = (m & 3) = (r & 3).
  #pragma unroll
  for (int t = 0; t < 8; ++t) {
    const int n = cg * 128 + t * 16 + ln;
    float b0, b1, b2, b3;
    if (biasPerDir) {
      b0 = bias[n]; b1 = bias[512 + n]; b2 = bias[1024 + n]; b3 = bias[1536 + n];
    } else {
      b0 = b1 = b2 = b3 = bias[n];
    }
    #pragma unroll
    for (int r = 0; r < 8; ++r) {
      const int m = rb + hi * 8 + r;
      float v = acc[t][r];
      const int d = r & 3;
      v += (d == 0) ? b0 : (d == 1) ? b1 : (d == 2) ? b2 : b3;
      if (Radd) v += (float)Radd[(size_t)(m >> 2) * IC + n];
      Y[(size_t)m * IC + n] = (bf16_t)v;
    }
  }
}

// ---------------------------------------------------------------------------
// 6) Output head: 512 -> 2 per row, one wave per row, shuffle reduction.
// ---------------------------------------------------------------------------
__global__ __launch_bounds__(256) void out_head(
    const bf16_t* __restrict__ X, const float* __restrict__ outW,
    const float* __restrict__ outb, float* __restrict__ Y)
{
  const int wid  = threadIdx.x >> 5;
  const int lane = threadIdx.x & 31;
  const int m = blockIdx.x * 8 + wid;   // grid sized so m < MROWS
  const bf16_t* xr = X + (size_t)m * IC + lane * 16;
  union { uint4 u[2]; __bf16 e[16]; } xv;
  xv.u[0] = *(const uint4*)(xr);
  xv.u[1] = *(const uint4*)(xr + 8);
  float s0 = 0.0f, s1 = 0.0f;
  #pragma unroll
  for (int i = 0; i < 16; ++i) {
    const int c = lane * 16 + i;
    const float x = (float)xv.e[i];
    s0 += x * outW[c * 2 + 0];
    s1 += x * outW[c * 2 + 1];
  }
  #pragma unroll
  for (int off = 16; off > 0; off >>= 1) {
    s0 += __shfl_xor(s0, off, 32);
    s1 += __shfl_xor(s1, off, 32);
  }
  if (lane == 0) {
    Y[(size_t)m * 2 + 0] = s0 + outb[0];
    Y[(size_t)m * 2 + 1] = s1 + outb[1];
  }
}

// ---------------------------------------------------------------------------
// Host launcher
// ---------------------------------------------------------------------------
extern "C" void kernel_launch(void* const* d_in, const int* in_sizes, int n_in,
                              void* d_out, int out_size, void* d_ws, size_t ws_size,
                              hipStream_t stream) {
  (void)in_sizes; (void)n_in; (void)out_size; (void)ws_size;

  const float* input_stuff = (const float*)d_in[0];
  const float* sound_loc   = (const float*)d_in[1];
  const float* grid_coors  = (const float*)d_in[2];
  const float* xy_offset   = (const float*)d_in[3];
  const float* bandwidths  = (const float*)d_in[4];
  const float* grid0       = (const float*)d_in[5];
  const float* proj_W      = (const float*)d_in[6];
  const float* proj_b      = (const float*)d_in[7];
  const float* res_W0      = (const float*)d_in[8];
  const float* res_b0      = (const float*)d_in[9];
  const float* res_W1      = (const float*)d_in[10];
  const float* res_b1      = (const float*)d_in[11];
  const float* layers_W    = (const float*)d_in[12];
  const float* layers_b    = (const float*)d_in[13];
  const float* channels    = (const float*)d_in[14];
  const float* out_W       = (const float*)d_in[15];
  const float* out_b       = (const float*)d_in[16];

  char* base = (char*)d_ws;
  size_t off = 0;
  auto alloc = [&](size_t bytes) -> char* {
    char* p = base + off;
    off = (off + bytes + 255) & ~(size_t)255;
    return p;
  };

  float*  tg       = (float*) alloc((size_t)B_ * 128 * 4);
  bf16_t* A0       = (bf16_t*)alloc((size_t)MBS * CATP * 2);
  bf16_t* projWt   = (bf16_t*)alloc((size_t)IC * CATP * 2);
  bf16_t* resW0t   = (bf16_t*)alloc((size_t)IC * CATP * 2);
  bf16_t* resW1t   = (bf16_t*)alloc((size_t)IC * IC * 2);
  bf16_t* layersWt = (bf16_t*)alloc((size_t)NLAY * IC * IC * 2);
  float*  bias0    = (float*) alloc((size_t)DIR * IC * 4);
  float*  biasK    = (float*) alloc((size_t)NLAY * DIR * IC * 4);
  bf16_t* T        = (bf16_t*)alloc((size_t)MBS * IC * 2);
  bf16_t* R        = (bf16_t*)alloc((size_t)MBS * IC * 2);
  bf16_t* Xa       = (bf16_t*)alloc((size_t)MROWS * IC * 2);
  bf16_t* Xb       = (bf16_t*)alloc((size_t)MROWS * IC * 2);

  // Front-end
  kr_kernel<<<16, 256, 0, stream>>>(sound_loc, grid_coors, xy_offset,
                                    bandwidths, grid0, tg);
  build_a0<<<(MBS * CATP + 255) / 256, 256, 0, stream>>>(tg, input_stuff, A0);

  const int twBlocks256 = (IC * CATP + 255) / 256;
  const int twBlocks512 = (IC * IC + 255) / 256;
  transpose_w<<<twBlocks256, 256, 0, stream>>>(proj_W, 248, CATP, projWt);
  transpose_w<<<twBlocks256, 256, 0, stream>>>(res_W0, 248, CATP, resW0t);
  transpose_w<<<twBlocks512, 256, 0, stream>>>(res_W1, IC, IC, resW1t);
  for (int k = 0; k < NLAY; ++k)
    transpose_w<<<twBlocks512, 256, 0, stream>>>(
        layers_W + (size_t)k * IC * IC, IC, IC, layersWt + (size_t)k * IC * IC);
  build_bias<<<((DIR + NLAY * DIR) * IC + 255) / 256, 256, 0, stream>>>(
      proj_b, layers_b, channels, bias0, biasK);

  // X0 = A0[m>>2] @ projWt + (proj_b + channels[0][d])   -> [64000,512]
  gemm512<<<MROWS / 32, 256, 0, stream>>>(A0, 1, CATP, projWt, bias0, 1,
                                          nullptr, 0, Xa);
  // Residual branch: T = A0 @ resW0 + b0 ; R = leaky(T) @ resW1 + b1
  gemm512<<<MBS / 32, 256, 0, stream>>>(A0, 0, CATP, resW0t, res_b0, 0,
                                        nullptr, 0, T);
  gemm512<<<MBS / 32, 256, 0, stream>>>(T, 0, IC, resW1t, res_b1, 0,
                                        nullptr, 1, R);

  // Main chain: X_{k+1} = leaky(X_k) @ W_k + b_k + ch_{k+1} (+R at k==2)
  bf16_t* cur = Xa;
  bf16_t* nxt = Xb;
  for (int k = 0; k < NLAY; ++k) {
    gemm512<<<MROWS / 32, 256, 0, stream>>>(
        cur, 0, IC, layersWt + (size_t)k * IC * IC, biasK + (size_t)k * DIR * IC,
        1, (k == 2) ? R : nullptr, 1, nxt);
    bf16_t* tmp = cur; cur = nxt; nxt = tmp;
  }

  // Head: [64000,512] @ [512,2] + out_b -> fp32 output
  out_head<<<MROWS / 8, 256, 0, stream>>>(cur, out_W, out_b, (float*)d_out);
}
